// CrystalDiffusionModel_8340826488880
// MI455X (gfx1250) — matrix operand order repaired and checked by reading, not compiled, not run on hardware
//
#include <hip/hip_runtime.h>
#include <hip/hip_bf16.h>
#include <math.h>

typedef __attribute__((ext_vector_type(16))) _Float16 v16h;
typedef __attribute__((ext_vector_type(8)))  _Float16 v8h;
typedef __attribute__((ext_vector_type(8)))  float    v8f;
typedef unsigned int u32x4 __attribute__((ext_vector_type(4)));
typedef int          i32x8 __attribute__((ext_vector_type(8)));
typedef int          i32x4 __attribute__((ext_vector_type(4)));

#define N_NODES 10000
#define N_EDGES 320000
#define EPRIME  (N_EDGES + N_NODES)   /* 330000, multiple of 16 */
#define HD      128
#define EC_WAVES 8                    /* waves (16-edge tiles) per edge block */

// Dynamic LDS layout for k_edge_conv (bytes)
#define LDS_W1_OFF   0
#define LDS_W2_OFF   65536
#define LDS_H1_OFF   98304                       /* 8 waves * 4KB */
#define LDS_DST_OFF  131072                      /* 8 waves * 16 ints */
#define LDS_TOTAL    (131072 + EC_WAVES * 16 * 4)

// ---------------------------------------------------------------------------
// TDM: DMA nBytes (multiple of 8, <=512KB) from global to LDS. One instruction;
// caller must s_wait_tensorcnt. D# packed per CDNA5 ISA ch.8 (2D tile, 1 row).
// ---------------------------------------------------------------------------
#if __has_builtin(__builtin_amdgcn_tensor_load_to_lds)
#define HAVE_TDM 1
__device__ inline void tdm_copy_to_lds(const void* gptr, unsigned lds_byte_off,
                                       unsigned nBytes) {
  unsigned L = nBytes >> 3;                       // elements of 8B
  unsigned long long ga = (unsigned long long)gptr;
  u32x4 g0;
  g0[0] = 1u;                                     // count=1 (valid), user mode
  g0[1] = lds_byte_off;                           // lds_addr [63:32]
  g0[2] = (unsigned)(ga & 0xffffffffu);           // global_addr [95:64]
  g0[3] = (unsigned)((ga >> 32) & 0x01ffffffu)    // global_addr [120:96]
          | (2u << 30);                           // type=2 ("image") [127:126]
  i32x8 g1;
  g1[0] = (int)(3u << 16);                        // wg_mask=0, data_size=3 (8B)
  g1[1] = (int)((L & 0xffffu) << 16);             // tensor_dim0 [15:0]
  g1[2] = (int)((L >> 16) & 0xffffu) | (1 << 16); // tensor_dim0 hi | tensor_dim1=1
  g1[3] = (int)((L & 0xffffu) << 16);             // tile_dim0 = L  (<= 65535)
  g1[4] = 1;                                      // tile_dim1=1, tile_dim2=0
  g1[5] = (int)L;                                 // tensor_dim0_stride lo32
  g1[6] = 0;
  g1[7] = 0;
  i32x4 z4 = {0, 0, 0, 0};
#if __clang_major__ >= 23
  i32x8 z8 = {0, 0, 0, 0, 0, 0, 0, 0};
  __builtin_amdgcn_tensor_load_to_lds(g0, g1, z4, z4, z8, 0);
#else
  __builtin_amdgcn_tensor_load_to_lds(g0, g1, z4, z4, 0);
#endif
}
#else
#define HAVE_TDM 0
#endif

// ---------------------------------------------------------------------------
// setup: cosine schedule scalars + sinusoidal time embedding -> t_emb[128]
// ---------------------------------------------------------------------------
__global__ void k_setup(const int* __restrict__ t_ptr,
                        const float* __restrict__ tp_w,
                        const float* __restrict__ tp_b,
                        float* __restrict__ scal,
                        float* __restrict__ t_emb) {
  int d = threadIdx.x;              // 128 threads
  float tf = (float)t_ptr[0] / 1000.0f;
  float acc = tp_b[d];
  for (int k = 0; k < 32; ++k) {
    float fr = expf(-4.0f + 8.0f * (float)k / 31.0f);
    float ph = tf * fr;
    acc += sinf(ph) * tp_w[k * HD + d];
    acc += cosf(ph) * tp_w[(32 + k) * HD + d];
  }
  t_emb[d] = acc;
  if (d == 0) {
    float s = 0.008f;
    float ang = (tf + s) / (1.0f + s) * (3.14159265358979323846f * 0.5f);
    float c = cosf(ang);
    float ab = c * c;
    scal[0] = sqrtf(ab);
    scal[1] = sqrtf(1.0f - ab);
  }
}

__global__ void k_h0(const float* __restrict__ x, const float* __restrict__ noise,
                     const float* __restrict__ scal, const float* __restrict__ t_emb,
                     float* __restrict__ h0) {
  int i = blockIdx.x * blockDim.x + threadIdx.x;
  if (i >= N_NODES * HD) return;
  h0[i] = scal[0] * x[i] + scal[1] * noise[i] + t_emb[i & (HD - 1)];
}

// Pack fp32 [K][N] weights into f16 WMMA B-fragment order (lane group g holds
// K = 8g..8g+7 then 16+8g..16+8g+7; per-lane 16 contiguous halves).
__global__ void k_pack(const float* __restrict__ W, _Float16* __restrict__ out,
                       int K, int N) {
  int idx = blockIdx.x * blockDim.x + threadIdx.x;
  if (idx >= K * N) return;
  int j    = idx & 15;
  int lane = (idx >> 4) & 31;
  int frag = idx >> 9;            // kt*NT + nt
  int NT = N >> 4;
  int nt = frag % NT;
  int kt = frag / NT;
  int g  = lane >> 4;
  int nl = lane & 15;
  int k  = kt * 32 + 8 * g + (j & 7) + ((j >> 3) << 4);
  int n  = nt * 16 + nl;
  out[idx] = (_Float16)W[k * N + n];
}

__global__ void k_fill_neg_inf(float* __restrict__ p, int n) {
  int i = blockIdx.x * blockDim.x + threadIdx.x;
  if (i < n) p[i] = -__builtin_huge_valf();
}

// sign-aware float max via integer atomics (monotone wrt IEEE ordering)
__device__ inline void atomicMaxF(float* addr, float v) {
  if (v >= 0.0f) atomicMax((int*)addr, __float_as_int(v));
  else           atomicMin((unsigned int*)addr, __float_as_uint(v));
}

// ---------------------------------------------------------------------------
// Node GEMM: out[N][128] = in[N][128] @ Wpk(f16,128x128) + b
// ---------------------------------------------------------------------------
__global__ __launch_bounds__(128, 1)
void k_node_gemm(const float* __restrict__ hin, const _Float16* __restrict__ wpk,
                 const float* __restrict__ bias, float* __restrict__ hout,
                 int nTiles) {
  int wave = blockIdx.x * (blockDim.x >> 5) + (threadIdx.x >> 5);
  if (wave >= nTiles) return;
  int lane = threadIdx.x & 31;
  int m = lane & 15, g = lane >> 4;
  const float* row = hin + (size_t)(wave * 16 + m) * HD;

  v8f acc[8] = {};
  #pragma unroll
  for (int c = 0; c < 4; ++c) {
    v16h Bf[8];
    #pragma unroll
    for (int nt = 0; nt < 8; ++nt)
      Bf[nt] = *(const v16h*)(wpk + (((size_t)(c * 8 + nt)) * 32 + lane) * 16);
    v16h A;
    const float* p0 = row + 32 * c + 8 * g;
    const float* p1 = row + 32 * c + 16 + 8 * g;
    #pragma unroll
    for (int j = 0; j < 8; ++j) { A[j] = (_Float16)p0[j]; A[8 + j] = (_Float16)p1[j]; }
    #pragma unroll
    for (int nt = 0; nt < 8; ++nt)
      acc[nt] = __builtin_amdgcn_wmma_f32_16x16x32_f16(false, A, false, Bf[nt],
                                                       (short)0, acc[nt], false, false);
  }
  #pragma unroll
  for (int nt = 0; nt < 8; ++nt) {
    int col = nt * 16 + m;
    float bv = bias[col];
    #pragma unroll
    for (int r = 0; r < 8; ++r) {
      int ro = wave * 16 + 8 * g + r;
      hout[(size_t)ro * HD + col] = acc[nt][r] + bv;
    }
  }
}

// ---------------------------------------------------------------------------
// EdgeConv (one wave per 16-edge tile, 8 waves/block, dynamic LDS):
//   block stages packed W1 (64KB) + W2 (32KB) in LDS via TDM, then per wave:
//   GEMM1 (K=256, B from LDS) -> relu -> f16 LDS tile
//   GEMM2 (K=128, B from LDS) -> bias -> atomic segment-max scatter
// ---------------------------------------------------------------------------
__global__ __launch_bounds__(256, 1)
void k_edge_conv(const float* __restrict__ hin, const int* __restrict__ ei,
                 const _Float16* __restrict__ w1pk, const float* __restrict__ b1,
                 const _Float16* __restrict__ w2pk, const float* __restrict__ b2,
                 float* __restrict__ hout, int nTiles) {
  extern __shared__ __attribute__((aligned(64))) char smem[];
  _Float16* sW1  = (_Float16*)(smem + LDS_W1_OFF);
  _Float16* sW2  = (_Float16*)(smem + LDS_W2_OFF);
  _Float16* sh1b = (_Float16*)(smem + LDS_H1_OFF);
  int*      sdsb = (int*)(smem + LDS_DST_OFF);

#if HAVE_TDM
  if (threadIdx.x == 0) {
    tdm_copy_to_lds(w1pk, (unsigned)(unsigned long long)(void*)sW1, 65536);
    tdm_copy_to_lds(w2pk, (unsigned)(unsigned long long)(void*)sW2, 32768);
    __builtin_amdgcn_s_wait_tensorcnt(0);
  }
#else
  for (unsigned i = threadIdx.x; i < 65536u / 8u; i += blockDim.x)
    ((unsigned long long*)sW1)[i] = ((const unsigned long long*)w1pk)[i];
  for (unsigned i = threadIdx.x; i < 32768u / 8u; i += blockDim.x)
    ((unsigned long long*)sW2)[i] = ((const unsigned long long*)w2pk)[i];
#endif
  __syncthreads();

  int wslot = threadIdx.x >> 5;
  int wave  = blockIdx.x * EC_WAVES + wslot;
  if (wave >= nTiles) return;
  int lane = threadIdx.x & 31;
  int m = lane & 15, g = lane >> 4;
  _Float16* tile = sh1b + wslot * 16 * HD;
  int*      sdst = sdsb + wslot * 16;

  int e = wave * 16 + m;
  int s, d;
  if (e < N_EDGES) { s = ei[e]; d = ei[N_EDGES + e]; }
  else             { s = e - N_EDGES; d = s; }       // self loops
  if (g == 0) sdst[m] = d;
  const float* xi = hin + (size_t)d * HD;
  const float* xj = hin + (size_t)s * HD;

  // ---- GEMM1: K = 256 (cols 0..127 = xi, 128..255 = xj - xi), B from LDS ----
  v8f acc[8] = {};
  #pragma unroll
  for (int c = 0; c < 8; ++c) {
    v16h Bf[8];
    #pragma unroll
    for (int nt = 0; nt < 8; ++nt)
      Bf[nt] = *(const v16h*)(sW1 + ((c * 8 + nt) * 32 + lane) * 16);
    v16h A;
    if (c < 4) {
      const float* p0 = xi + 32 * c + 8 * g;
      const float* p1 = xi + 32 * c + 16 + 8 * g;
      #pragma unroll
      for (int j = 0; j < 8; ++j) { A[j] = (_Float16)p0[j]; A[8 + j] = (_Float16)p1[j]; }
    } else {
      int b0 = 32 * (c - 4) + 8 * g, b1o = 32 * (c - 4) + 16 + 8 * g;
      #pragma unroll
      for (int j = 0; j < 8; ++j) {
        A[j]     = (_Float16)(xj[b0 + j]  - xi[b0 + j]);
        A[8 + j] = (_Float16)(xj[b1o + j] - xi[b1o + j]);
      }
    }
    #pragma unroll
    for (int nt = 0; nt < 8; ++nt)
      acc[nt] = __builtin_amdgcn_wmma_f32_16x16x32_f16(false, A, false, Bf[nt],
                                                       (short)0, acc[nt], false, false);
  }

  // bias + ReLU -> f16 LDS tile, row-major 16x128 (cross-lane re-layout)
  #pragma unroll
  for (int nt = 0; nt < 8; ++nt) {
    int col = nt * 16 + m;
    float bv = b1[col];
    #pragma unroll
    for (int r = 0; r < 8; ++r) {
      float v = acc[nt][r] + bv;
      v = v > 0.0f ? v : 0.0f;
      tile[(8 * g + r) * HD + col] = (_Float16)v;
    }
  }
  asm volatile("s_wait_dscnt 0x0" ::: "memory");   // per-wave LDS RAW fence

  // ---- GEMM2: K = 128, B fragments from LDS ----
  v8f acc2[8] = {};
  #pragma unroll
  for (int c = 0; c < 4; ++c) {
    v16h Bf[8];
    #pragma unroll
    for (int nt = 0; nt < 8; ++nt)
      Bf[nt] = *(const v16h*)(sW2 + ((c * 8 + nt) * 32 + lane) * 16);
    v8h lo = *(const v8h*)(tile + m * HD + 32 * c + 8 * g);
    v8h hi = *(const v8h*)(tile + m * HD + 32 * c + 16 + 8 * g);
    v16h A;
    #pragma unroll
    for (int j = 0; j < 8; ++j) { A[j] = lo[j]; A[8 + j] = hi[j]; }
    #pragma unroll
    for (int nt = 0; nt < 8; ++nt)
      acc2[nt] = __builtin_amdgcn_wmma_f32_16x16x32_f16(false, A, false, Bf[nt],
                                                        (short)0, acc2[nt], false, false);
  }

  // bias + scatter segment-max
  #pragma unroll
  for (int nt = 0; nt < 8; ++nt) {
    int col = nt * 16 + m;
    float bv = b2[col];
    #pragma unroll
    for (int r = 0; r < 8; ++r) {
      float v  = acc2[nt][r] + bv;
      int   dn = sdst[8 * g + r];
      atomicMaxF(hout + (size_t)dn * HD + col, v);
    }
  }
}

// ---------------------------------------------------------------------------
// loss = mean((score - noise)^2)
// ---------------------------------------------------------------------------
__global__ void k_zero(float* out) { out[0] = 0.0f; }

__global__ void k_loss(const float* __restrict__ score, const float* __restrict__ noise,
                       float* __restrict__ out) {
  __shared__ float red[256];
  float v = 0.0f;
  for (int i = blockIdx.x * blockDim.x + threadIdx.x; i < N_NODES * HD;
       i += gridDim.x * blockDim.x) {
    float dlt = score[i] - noise[i];
    v += dlt * dlt;
  }
  red[threadIdx.x] = v;
  __syncthreads();
  for (int s = 128; s > 0; s >>= 1) {
    if (threadIdx.x < s) red[threadIdx.x] += red[threadIdx.x + s];
    __syncthreads();
  }
  if (threadIdx.x == 0) atomicAdd(out, red[0] * (1.0f / (float)(N_NODES * HD)));
}

// ---------------------------------------------------------------------------
extern "C" void kernel_launch(void* const* d_in, const int* in_sizes, int n_in,
                              void* d_out, int out_size, void* d_ws, size_t ws_size,
                              hipStream_t stream) {
  (void)in_sizes; (void)n_in; (void)out_size; (void)ws_size;
  const float* x       = (const float*)d_in[0];
  const float* noise   = (const float*)d_in[1];
  const int*   ei      = (const int*)d_in[2];
  const int*   t       = (const int*)d_in[3];
  const float* emb_w   = (const float*)d_in[4];
  const float* emb_b   = (const float*)d_in[5];
  const float* enc_w1  = (const float*)d_in[6];
  const float* enc_b1  = (const float*)d_in[7];
  const float* enc_w2  = (const float*)d_in[8];
  const float* enc_b2  = (const float*)d_in[9];
  const float* encfc_w = (const float*)d_in[10];
  const float* encfc_b = (const float*)d_in[11];
  const float* dec_w1  = (const float*)d_in[12];
  const float* dec_b1  = (const float*)d_in[13];
  const float* dec_w2  = (const float*)d_in[14];
  const float* dec_b2  = (const float*)d_in[15];
  const float* decfc_w = (const float*)d_in[16];
  const float* decfc_b = (const float*)d_in[17];
  const float* tp_w    = (const float*)d_in[18];
  const float* tp_b    = (const float*)d_in[19];

  char* ws = (char*)d_ws;
  float*    scal  = (float*)(ws + 0);
  float*    t_emb = (float*)(ws + 256);
  _Float16* wpk   = (_Float16*)(ws + 1024);

  const size_t SZ128 = 128 * 128;   // halves per packed 128x128
  const size_t SZ256 = 256 * 128;   // halves per packed 256x128
  _Float16* p_emb = wpk;
  _Float16* p_ew1 = p_emb + SZ128;
  _Float16* p_ew2 = p_ew1 + 4 * SZ256;
  _Float16* p_efc = p_ew2 + 4 * SZ128;
  _Float16* p_dw1 = p_efc + SZ128;
  _Float16* p_dw2 = p_dw1 + 4 * SZ256;
  _Float16* p_dfc = p_dw2 + 4 * SZ128;
  size_t wpk_bytes = (size_t)((p_dfc + SZ128) - wpk) * sizeof(_Float16);
  size_t hoff = (1024 + wpk_bytes + 1023) & ~(size_t)1023;
  float* hA = (float*)(ws + hoff);
  float* hB = hA + (size_t)N_NODES * HD;

  k_setup<<<1, 128, 0, stream>>>(t, tp_w, tp_b, scal, t_emb);

  auto pack = [&](const float* W, _Float16* out, int K, int Ncols) {
    int tot = K * Ncols;
    k_pack<<<(tot + 255) / 256, 256, 0, stream>>>(W, out, K, Ncols);
  };
  pack(emb_w, p_emb, 128, 128);
  for (int i = 0; i < 4; ++i) {
    pack(enc_w1 + (size_t)i * 256 * 128, p_ew1 + (size_t)i * SZ256, 256, 128);
    pack(enc_w2 + (size_t)i * 128 * 128, p_ew2 + (size_t)i * SZ128, 128, 128);
    pack(dec_w1 + (size_t)i * 256 * 128, p_dw1 + (size_t)i * SZ256, 256, 128);
    pack(dec_w2 + (size_t)i * 128 * 128, p_dw2 + (size_t)i * SZ128, 128, 128);
  }
  pack(encfc_w, p_efc, 128, 128);
  pack(decfc_w, p_dfc, 128, 128);

  k_h0<<<(N_NODES * HD + 255) / 256, 256, 0, stream>>>(x, noise, scal, t_emb, hA);

  const int nodeTiles  = N_NODES / 16;                 // 625
  const int nodeBlocks = (nodeTiles + 3) / 4;
  const int edgeTiles  = EPRIME / 16;                  // 20625
  const int edgeBlocks = (edgeTiles + EC_WAVES - 1) / EC_WAVES;
  const int fillBlocks = (N_NODES * HD + 255) / 256;

  k_node_gemm<<<nodeBlocks, 128, 0, stream>>>(hA, p_emb, emb_b, hB, nodeTiles);
  float* cur = hB; float* other = hA;

  for (int i = 0; i < 4; ++i) {   // encoder EdgeConvs
    k_fill_neg_inf<<<fillBlocks, 256, 0, stream>>>(other, N_NODES * HD);
    k_edge_conv<<<edgeBlocks, 256, LDS_TOTAL, stream>>>(cur, ei,
        p_ew1 + (size_t)i * SZ256, enc_b1 + i * 128,
        p_ew2 + (size_t)i * SZ128, enc_b2 + i * 128, other, edgeTiles);
    float* tmp = cur; cur = other; other = tmp;
  }
  k_node_gemm<<<nodeBlocks, 128, 0, stream>>>(cur, p_efc, encfc_b, other, nodeTiles);
  { float* tmp = cur; cur = other; other = tmp; }
  for (int i = 0; i < 4; ++i) {   // decoder EdgeConvs
    k_fill_neg_inf<<<fillBlocks, 256, 0, stream>>>(other, N_NODES * HD);
    k_edge_conv<<<edgeBlocks, 256, LDS_TOTAL, stream>>>(cur, ei,
        p_dw1 + (size_t)i * SZ256, dec_b1 + i * 128,
        p_dw2 + (size_t)i * SZ128, dec_b2 + i * 128, other, edgeTiles);
    float* tmp = cur; cur = other; other = tmp;
  }
  k_node_gemm<<<nodeBlocks, 128, 0, stream>>>(cur, p_dfc, decfc_b, other, nodeTiles);
  cur = other;

  k_zero<<<1, 1, 0, stream>>>((float*)d_out);
  k_loss<<<256, 256, 0, stream>>>(cur, noise, (float*)d_out);
}
